// STAttention_65781719105748
// MI455X (gfx1250) — matrix operand order repaired and checked by reading, not compiled, not run on hardware
//
#include <hip/hip_runtime.h>

// ---------------- types ----------------
typedef __bf16 bf16;
typedef __attribute__((ext_vector_type(16))) __bf16 v16bf;
typedef __attribute__((ext_vector_type(8)))  __bf16 v8bf;
typedef __attribute__((ext_vector_type(8)))  float  v8f;

// ---------------- problem constants ----------------
#define BB   8
#define TT   24
#define NN   263
#define DD   512
#define N0_  32
#define N1_  64
#define SEG_ 12
#define STW  640               // concat width: 256 s_x | 128 sg | 128 t_x | 128 tg
#define NPADS 272              // 263 padded to 16
#define KPADS 288              // 272 padded to 32 for PV k-loop
#define LAMBDA_INIT_F 0.2f     // 0.8 - 0.6*exp(0)

// ---------------- WMMA helpers ----------------
__device__ __forceinline__ v8f v8f_zero() {
  v8f z;
#pragma unroll
  for (int i = 0; i < 8; ++i) z[i] = 0.f;
  return z;
}

__device__ __forceinline__ v8f wmma_bf16(v16bf a, v16bf b, v8f c) {
  // D = A(16x32 bf16) x B(32x16 bf16) + C(16x16 f32)
  return __builtin_amdgcn_wmma_f32_16x16x32_bf16(false, a, false, b, (short)0, c,
                                                 false, false);
}

// A-matrix 16x32 tile from row-major memory.
// lanes 0-15 : M=lane,   K {k0..k0+15}
// lanes 16-31: M=lane-16,K {k0+16..k0+31}
__device__ __forceinline__ v16bf load_A_g(const bf16* __restrict__ A, int lda,
                                          int row0, int k0, int rowmax) {
  const int lane = threadIdx.x & 31, r = lane & 15, half = lane >> 4;
  int row = row0 + r; row = row > rowmax ? rowmax : row;
  const bf16* p = A + (size_t)row * lda + k0 + half * 16;
  v8bf lo = *(const v8bf*)p;
  v8bf hi = *(const v8bf*)(p + 8);
  v16bf o;
#pragma unroll
  for (int i = 0; i < 8; ++i) { o[i] = lo[i]; o[i + 8] = hi[i]; }
  return o;
}

__device__ __forceinline__ v16bf load_A_lds(const bf16* P, int stride, int row0, int k0) {
  const int lane = threadIdx.x & 31, r = lane & 15, half = lane >> 4;
  const bf16* p = P + (size_t)(row0 + r) * stride + k0 + half * 16;
  v8bf lo = *(const v8bf*)p;
  v8bf hi = *(const v8bf*)(p + 8);
  v16bf o;
#pragma unroll
  for (int i = 0; i < 8; ++i) { o[i] = lo[i]; o[i + 8] = hi[i]; }
  return o;
}

// B-matrix 32x16 tile where B[k][col] = W[col][k] with W row-major ExK
// (columns of B are contiguous rows of W).
__device__ __forceinline__ v16bf load_B_rows(const bf16* __restrict__ W, int ldw,
                                             int col0, int k0, int colmax) {
  const int lane = threadIdx.x & 31, r = lane & 15, half = lane >> 4;
  int col = col0 + r; col = col > colmax ? colmax : col;
  const bf16* p = W + (size_t)col * ldw + k0 + half * 16;
  v8bf lo = *(const v8bf*)p;
  v8bf hi = *(const v8bf*)(p + 8);
  v16bf o;
#pragma unroll
  for (int i = 0; i < 8; ++i) { o[i] = lo[i]; o[i + 8] = hi[i]; }
  return o;
}

// B-matrix 32x16 tile from a row-major KxN buffer (e.g. V): strided loads.
__device__ __forceinline__ v16bf load_B_strided(const bf16* __restrict__ Bm, int ldb,
                                                int k0, int col0, int kmax) {
  const int lane = threadIdx.x & 31, r = lane & 15, half = lane >> 4;
  v16bf o;
#pragma unroll
  for (int j = 0; j < 16; ++j) {
    int k = k0 + half * 16 + j; k = k > kmax ? kmax : k;
    o[j] = Bm[(size_t)k * ldb + col0 + r];
  }
  return o;
}

// B-matrix 32x16 tile from a 32-row LDS panel (strided LDS reads).
__device__ __forceinline__ v16bf load_B_lds(const bf16* Bm, int ldb, int col0) {
  const int lane = threadIdx.x & 31, r = lane & 15, half = lane >> 4;
  v16bf o;
#pragma unroll
  for (int j = 0; j < 16; ++j)
    o[j] = Bm[(size_t)(half * 16 + j) * ldb + col0 + r];
  return o;
}

// ---------------- utility kernels ----------------
__global__ void cast_f32_to_bf16(const float* __restrict__ in, bf16* __restrict__ out,
                                 size_t n) {
  size_t i = (size_t)blockIdx.x * blockDim.x + threadIdx.x;
  size_t stride = (size_t)gridDim.x * blockDim.x;
  for (; i < n; i += stride) out[i] = (bf16)in[i];
}

__global__ void lam_kernel(const float* __restrict__ lq1, const float* __restrict__ lk1,
                           const float* __restrict__ lq2, const float* __restrict__ lk2,
                           float* __restrict__ lam) {
  if (threadIdx.x == 0) {
    float s1 = 0.f, s2 = 0.f;
    for (int i = 0; i < 32; ++i) { s1 += lq1[i] * lk1[i]; s2 += lq2[i] * lk2[i]; }
    *lam = __expf(s1) - __expf(s2) + LAMBDA_INIT_F;
  }
}

// ---------------- generic GEMM: C[M,E] = A[M,K] * W[E,K]^T (+bias) ----------------
// one wave computes a 32x64 tile (2 A-tiles share 4 B-tiles -> 8 WMMA per K-step)
// grid = (M/32, E/64)
__global__ void __launch_bounds__(32)
wmma_gemm_kernel(const bf16* __restrict__ A, int lda,
                 const bf16* __restrict__ W, int ldw,
                 const float* __restrict__ bias,
                 void* __restrict__ Cout, int ldc, int col_off, int c_is_bf16,
                 int M, int E, int K) {
  const int tm = blockIdx.x * 32;
  const int te = blockIdx.y * 64;
  v8f acc[2][4];
#pragma unroll
  for (int u = 0; u < 2; ++u)
#pragma unroll
    for (int j = 0; j < 4; ++j) acc[u][j] = v8f_zero();

  for (int k0 = 0; k0 < K; k0 += 32) {
    if (k0 + 32 < K) {
      __builtin_prefetch(A + (size_t)tm * lda + k0 + 32, 0, 3);        // near prefetch
      __builtin_prefetch(A + (size_t)(tm + 16) * lda + k0 + 32, 0, 3);
    }
    v16bf a0 = load_A_g(A, lda, tm, k0, M - 1);
    v16bf a1 = load_A_g(A, lda, tm + 16, k0, M - 1);
#pragma unroll
    for (int j = 0; j < 4; ++j) {
      v16bf b = load_B_rows(W, ldw, te + j * 16, k0, E - 1);
      acc[0][j] = wmma_bf16(a0, b, acc[0][j]);
      acc[1][j] = wmma_bf16(a1, b, acc[1][j]);
    }
  }
  const int lane = threadIdx.x & 31, r = lane & 15, half = lane >> 4;
#pragma unroll
  for (int u = 0; u < 2; ++u)
#pragma unroll
    for (int j = 0; j < 4; ++j)
#pragma unroll
      for (int i = 0; i < 8; ++i) {
        int row = tm + u * 16 + i + half * 8;
        int col = te + j * 16 + r;
        if (row < M && col < E) {
          float v = acc[u][j][i] + (bias ? bias[col] : 0.f);
          if (c_is_bf16) ((bf16*)Cout)[(size_t)row * ldc + col_off + col] = (bf16)v;
          else           ((float*)Cout)[(size_t)row * ldc + col_off + col] = v;
        }
      }
}

// ---------------- differential spatial attention (s_x) ----------------
// one wave per (row-tile, head, bt). Computes P for both lambda variants first,
// then a single PV K-loop stages V chunks into LDS (contiguous loads) and feeds
// both accumulators; combines with lambda, LayerNorms over the 64-wide head and
// writes bf16 into st[:, h*64 .. h*64+63].
__global__ void __launch_bounds__(32)
diff_attn_kernel(const bf16* __restrict__ q, const bf16* __restrict__ k,
                 const bf16* __restrict__ v, const float* __restrict__ lam_p,
                 const float* __restrict__ ln_g, const float* __restrict__ ln_b,
                 bf16* __restrict__ st) {
  __shared__ float sS[16][NPADS];          // raw scores / exp workspace
  __shared__ bf16  sP[2][16][KPADS];       // normalized P, one per lambda-variant
  __shared__ bf16  sV[32][64];             // staged V chunk (32 tokens x 64 cols)
  __shared__ float sO[16][64];             // combined head output
  const int rt = blockIdx.x;   // 0..16
  const int h  = blockIdx.y;   // 0..3
  const int bt = blockIdx.z;   // 0..191
  const int lane = threadIdx.x & 31, r = lane & 15, half = lane >> 4;
  const int row0 = rt * 16;
  const size_t base = (size_t)bt * NN * 256;
  const float sc = 0.17677669529663687f;  // 32^-0.5
  const float lam = *lam_p;

  // ---- scores + softmax for both variants ----
  for (int var = 0; var < 2; ++var) {
    const bf16* qb = q + base + h * 64 + var * 32;
    const bf16* kb = k + base + h * 64 + var * 32;
    v16bf a = load_A_g(qb, 256, row0, 0, NN - 1);   // dh=32 -> one WMMA per m-tile
    for (int mt = 0; mt < 17; ++mt) {
      v16bf b = load_B_rows(kb, 256, mt * 16, 0, NN - 1);
      v8f c = v8f_zero();
      c = wmma_bf16(a, b, c);
#pragma unroll
      for (int i = 0; i < 8; ++i) {
        int m = mt * 16 + r;
        sS[i + half * 8][m] = (m < NN) ? c[i] * sc : -1e30f;
      }
    }
    // row softmax (2 lanes per row)
    float mx = -1e30f;
    for (int c0 = half; c0 < NPADS; c0 += 2) mx = fmaxf(mx, sS[r][c0]);
    mx = fmaxf(mx, __shfl_xor(mx, 16, 32));
    float sum = 0.f;
    for (int c0 = half; c0 < NPADS; c0 += 2) {
      float e = __expf(sS[r][c0] - mx); sS[r][c0] = e; sum += e;
    }
    sum += __shfl_xor(sum, 16, 32);
    float inv = 1.f / sum;
    for (int c0 = half; c0 < NPADS; c0 += 2) sP[var][r][c0] = (bf16)(sS[r][c0] * inv);
    for (int c0 = NPADS + half; c0 < KPADS; c0 += 2) sP[var][r][c0] = (bf16)0.f;
  }

  // ---- PV for both variants, V chunk staged once per K-step ----
  v8f o1[4], o2[4];
#pragma unroll
  for (int j = 0; j < 4; ++j) { o1[j] = v8f_zero(); o2[j] = v8f_zero(); }
  const bf16* vb = v + base + h * 64;
  for (int kc = 0; kc < KPADS; kc += 32) {
    // stage 32 token rows x 64 cols of V into LDS with contiguous 16B loads
    int tok = kc + lane; tok = tok > NN - 1 ? NN - 1 : tok;
    const bf16* vrow = vb + (size_t)tok * 256;
#pragma unroll
    for (int c8 = 0; c8 < 8; ++c8)
      *(v8bf*)&sV[lane][c8 * 8] = *(const v8bf*)(vrow + c8 * 8);
    v16bf pa1 = load_A_lds(&sP[0][0][0], KPADS, 0, kc);
    v16bf pa2 = load_A_lds(&sP[1][0][0], KPADS, 0, kc);
#pragma unroll
    for (int j = 0; j < 4; ++j) {
      v16bf bv = load_B_lds(&sV[0][0], 64, j * 16);
      o1[j] = wmma_bf16(pa1, bv, o1[j]);
      o2[j] = wmma_bf16(pa2, bv, o2[j]);
    }
  }
  // combine with lambda
#pragma unroll
  for (int j = 0; j < 4; ++j)
#pragma unroll
    for (int i = 0; i < 8; ++i)
      sO[i + half * 8][j * 16 + r] = o1[j][i] - lam * o2[j][i];

  // ---- LayerNorm over 64 + * (1-LAMBDA_INIT), bf16 store into st ----
  float s1 = 0.f, s2 = 0.f;
  for (int c0 = half * 32; c0 < half * 32 + 32; ++c0) {
    float x = sO[r][c0]; s1 += x; s2 += x * x;
  }
  s1 += __shfl_xor(s1, 16, 32);
  s2 += __shfl_xor(s2, 16, 32);
  float mean = s1 * (1.f / 64.f);
  float varr = s2 * (1.f / 64.f) - mean * mean;
  float rstd = rsqrtf(varr + 1e-5f);
  int row = row0 + r;
  if (row < NN) {
    bf16* dst = st + ((size_t)bt * NN + row) * STW + h * 64;
    for (int c0 = half * 32; c0 < half * 32 + 32; ++c0) {
      float xv = (sO[r][c0] - mean) * rstd * ln_g[c0] + ln_b[c0];
      dst[c0] = (bf16)(xv * (1.f - LAMBDA_INIT_F));
    }
  }
}

// ---------------- temporal attention (t_x): 24x24 per (b,n,h) ----------------
__global__ void __launch_bounds__(32)
temporal_attn_kernel(const bf16* __restrict__ q, const bf16* __restrict__ k,
                     const bf16* __restrict__ v, bf16* __restrict__ st, int col_base) {
  __shared__ float sS[32][32];
  __shared__ bf16  sP[32][32];
  const int n = blockIdx.x, h = blockIdx.y, b = blockIdx.z;
  const int lane = threadIdx.x & 31, r = lane & 15, half = lane >> 4;
  const int ld = NN * 128;                       // stride between t rows
  const size_t base = ((size_t)b * TT * NN + n) * 128 + h * 64;
  const float sc = 0.125f;                       // 64^-0.5
  for (int rt = 0; rt < 2; ++rt)
    for (int ct = 0; ct < 2; ++ct) {
      v8f c = v8f_zero();
#pragma unroll
      for (int kc = 0; kc < 64; kc += 32) {
        v16bf a  = load_A_g(q + base, ld, rt * 16, kc, TT - 1);
        v16bf bb = load_B_rows(k + base, ld, ct * 16, kc, TT - 1);
        c = wmma_bf16(a, bb, c);
      }
#pragma unroll
      for (int i = 0; i < 8; ++i) {
        int cc = ct * 16 + r;
        sS[rt * 16 + i + half * 8][cc] = (cc < TT) ? c[i] * sc : -1e30f;
      }
    }
  {  // softmax: one row per lane (32 rows)
    int row = lane;
    float mx = -1e30f;
    for (int c0 = 0; c0 < TT; ++c0) mx = fmaxf(mx, sS[row][c0]);
    float sum = 0.f;
    for (int c0 = 0; c0 < 32; ++c0) { float e = __expf(sS[row][c0] - mx); sS[row][c0] = e; sum += e; }
    float inv = 1.f / sum;
    for (int c0 = 0; c0 < 32; ++c0) sP[row][c0] = (bf16)(sS[row][c0] * inv);
  }
  for (int rt = 0; rt < 2; ++rt) {
    v8f o[4];
#pragma unroll
    for (int j = 0; j < 4; ++j) o[j] = v8f_zero();
    v16bf pa = load_A_lds(&sP[0][0], 32, rt * 16, 0);
#pragma unroll
    for (int j = 0; j < 4; ++j) {
      v16bf bv = load_B_strided(v + base, ld, 0, j * 16, TT - 1);
      o[j] = wmma_bf16(pa, bv, o[j]);
    }
#pragma unroll
    for (int j = 0; j < 4; ++j)
#pragma unroll
      for (int i = 0; i < 8; ++i) {
        int t = rt * 16 + i + half * 8;
        if (t < TT)
          st[(((size_t)b * TT + t) * NN + n) * STW + col_base + h * 64 + j * 16 + r] =
              (bf16)o[j][i];
      }
  }
}

// ---------------- temporal-global attention: qg (12) x kg (24) per (b,n,h) ----------------
__global__ void __launch_bounds__(32)
tglobal_attn_kernel(const bf16* __restrict__ qg, const bf16* __restrict__ kg,
                    const bf16* __restrict__ vg, float* __restrict__ tgx) {
  __shared__ float sS[16][32];
  __shared__ bf16  sP[16][32];
  const int n = blockIdx.x, h = blockIdx.y, b = blockIdx.z;
  const int lane = threadIdx.x & 31, r = lane & 15, half = lane >> 4;
  const int ld = NN * 128;
  const size_t kbase = ((size_t)b * TT * NN + n) * 128 + h * 64;
  const size_t qbase = (size_t)n * SEG_ * 128 + h * 64;
  const float sc = 0.125f;
  for (int ct = 0; ct < 2; ++ct) {
    v8f c = v8f_zero();
#pragma unroll
    for (int kc = 0; kc < 64; kc += 32) {
      v16bf a  = load_A_g(qg + qbase, 128, 0, kc, SEG_ - 1);
      v16bf bb = load_B_rows(kg + kbase, ld, ct * 16, kc, TT - 1);
      c = wmma_bf16(a, bb, c);
    }
#pragma unroll
    for (int i = 0; i < 8; ++i) {
      int s = i + half * 8, t = ct * 16 + r;
      sS[s][t] = (t >= TT) ? -1e30f : (s >= SEG_ ? 0.f : c[i] * sc);
    }
  }
  float mx = -1e30f;
  for (int c0 = half; c0 < 32; c0 += 2) mx = fmaxf(mx, sS[r][c0]);
  mx = fmaxf(mx, __shfl_xor(mx, 16, 32));
  float sum = 0.f;
  for (int c0 = half; c0 < 32; c0 += 2) { float e = __expf(sS[r][c0] - mx); sS[r][c0] = e; sum += e; }
  sum += __shfl_xor(sum, 16, 32);
  float inv = 1.f / sum;
  for (int c0 = half; c0 < 32; c0 += 2) sP[r][c0] = (bf16)(sS[r][c0] * inv);
  v8f o[4];
#pragma unroll
  for (int j = 0; j < 4; ++j) o[j] = v8f_zero();
  v16bf pa = load_A_lds(&sP[0][0], 32, 0, 0);
#pragma unroll
  for (int j = 0; j < 4; ++j) {
    v16bf bv = load_B_strided(vg + kbase, ld, 0, j * 16, TT - 1);
    o[j] = wmma_bf16(pa, bv, o[j]);
  }
#pragma unroll
  for (int j = 0; j < 4; ++j)
#pragma unroll
    for (int i = 0; i < 8; ++i) {
      int s = i + half * 8;
      if (s < SEG_)
        tgx[(((size_t)b * NN + n) * SEG_ + s) * 128 + h * 64 + j * 16 + r] = o[j][i];
    }
}

// tg[b,t,n,c] = sum_s tmp_map[b,n,t,s] * tgx[b,n,s,c] -> st[:, 512+c]
__global__ void tgmap_kernel(const float* __restrict__ tmp_map,
                             const float* __restrict__ tgx, bf16* __restrict__ st) {
  size_t idx = (size_t)blockIdx.x * blockDim.x + threadIdx.x;
  const size_t total = (size_t)BB * TT * NN * 128;
  if (idx >= total) return;
  int c = (int)(idx & 127);
  size_t btn = idx >> 7;
  int n = (int)(btn % NN);
  size_t bt = btn / NN;
  int t = (int)(bt % TT);
  int b = (int)(bt / TT);
  const float* tm = tmp_map + (((size_t)b * NN + n) * TT + t) * SEG_;
  const float* tx = tgx + ((size_t)b * NN + n) * SEG_ * 128 + c;
  float acc = 0.f;
#pragma unroll
  for (int s = 0; s < SEG_; ++s) acc += tm[s] * tx[(size_t)s * 128];
  st[btn * STW + 512 + c] = (bf16)acc;
}

// ---------------- aggregation-branch attention: Ni x Ni per (bt,h,rowtile) ----------------
__global__ void __launch_bounds__(32)
agg_attn_kernel(const bf16* __restrict__ q, const bf16* __restrict__ k,
                const bf16* __restrict__ v, float* __restrict__ sx, int Ni) {
  __shared__ float sS[16][64];
  __shared__ bf16  sP[16][64];
  const int rt = blockIdx.x, h = blockIdx.y, bt = blockIdx.z;
  const int lane = threadIdx.x & 31, r = lane & 15, half = lane >> 4;
  const size_t base = (size_t)bt * Ni * 128 + h * 64;
  const float sc = 0.125f;
  const int row0 = rt * 16;
  const int nct = Ni / 16, nkc = Ni / 32;
  for (int ct = 0; ct < nct; ++ct) {
    v8f c = v8f_zero();
#pragma unroll
    for (int kc = 0; kc < 64; kc += 32) {
      v16bf a  = load_A_g(q + base, 128, row0, kc, Ni - 1);
      v16bf bb = load_B_rows(k + base, 128, ct * 16, kc, Ni - 1);
      c = wmma_bf16(a, bb, c);
    }
#pragma unroll
    for (int i = 0; i < 8; ++i) sS[i + half * 8][ct * 16 + r] = c[i] * sc;
  }
  float mx = -1e30f;
  for (int c0 = half; c0 < Ni; c0 += 2) mx = fmaxf(mx, sS[r][c0]);
  mx = fmaxf(mx, __shfl_xor(mx, 16, 32));
  float sum = 0.f;
  for (int c0 = half; c0 < Ni; c0 += 2) { float e = __expf(sS[r][c0] - mx); sS[r][c0] = e; sum += e; }
  sum += __shfl_xor(sum, 16, 32);
  float inv = 1.f / sum;
  for (int c0 = half; c0 < Ni; c0 += 2) sP[r][c0] = (bf16)(sS[r][c0] * inv);
  v8f o[4];
#pragma unroll
  for (int j = 0; j < 4; ++j) o[j] = v8f_zero();
  for (int kc = 0; kc < nkc; ++kc) {
    v16bf pa = load_A_lds(&sP[0][0], 64, 0, kc * 32);
#pragma unroll
    for (int j = 0; j < 4; ++j) {
      v16bf bv = load_B_strided(v + base, 128, kc * 32, j * 16, Ni - 1);
      o[j] = wmma_bf16(pa, bv, o[j]);
    }
  }
#pragma unroll
  for (int j = 0; j < 4; ++j)
#pragma unroll
    for (int i = 0; i < 8; ++i) {
      int m = row0 + i + half * 8;
      sx[((size_t)bt * Ni + m) * 128 + h * 64 + j * 16 + r] = o[j][i];
    }
}

// sg[bt,n,c] = sum_m M0[m,n]*sx0[bt,m,c] + sum_m M1[m,n]*sx1[bt,m,c]
__global__ void mproj_kernel(const float* __restrict__ sx0, const float* __restrict__ M0w,
                             const float* __restrict__ sx1, const float* __restrict__ M1w,
                             float* __restrict__ sg) {
  size_t idx = (size_t)blockIdx.x * blockDim.x + threadIdx.x;
  const size_t total = (size_t)BB * TT * NN * 128;
  if (idx >= total) return;
  int c = (int)(idx & 127);
  size_t btn = idx >> 7;
  int n = (int)(btn % NN);
  size_t bt = btn / NN;
  float acc = 0.f;
  const float* s0 = sx0 + (size_t)bt * N0_ * 128 + c;
  for (int m = 0; m < N0_; ++m) acc += M0w[(size_t)m * NN + n] * s0[(size_t)m * 128];
  const float* s1 = sx1 + (size_t)bt * N1_ * 128 + c;
  for (int m = 0; m < N1_; ++m) acc += M1w[(size_t)m * NN + n] * s1[(size_t)m * 128];
  sg[idx] = acc;
}

// ---------------- host orchestration ----------------
extern "C" void kernel_launch(void* const* d_in, const int* in_sizes, int n_in,
                              void* d_out, int out_size, void* d_ws, size_t ws_size,
                              hipStream_t stream) {
  (void)in_sizes; (void)n_in; (void)out_size; (void)ws_size;
  const float* x    = (const float*)d_in[0];
  const float* ax0  = (const float*)d_in[1];
  const float* ax1  = (const float*)d_in[2];
  const float* tmap = (const float*)d_in[3];
  const float* Wq_s = (const float*)d_in[4];
  const float* Wk_s = (const float*)d_in[5];
  const float* Wv_s = (const float*)d_in[6];
  const float* lq1  = (const float*)d_in[7];
  const float* lk1  = (const float*)d_in[8];
  const float* lq2  = (const float*)d_in[9];
  const float* lk2  = (const float*)d_in[10];
  const float* ln_g = (const float*)d_in[11];
  const float* ln_b = (const float*)d_in[12];
  const float* Wq_a0 = (const float*)d_in[13];
  const float* Wk_a0 = (const float*)d_in[14];
  const float* Wv_a0 = (const float*)d_in[15];
  const float* Wq_a1 = (const float*)d_in[16];
  const float* Wk_a1 = (const float*)d_in[17];
  const float* Wv_a1 = (const float*)d_in[18];
  const float* M0w  = (const float*)d_in[19];
  const float* M1w  = (const float*)d_in[20];
  const float* Wagg = (const float*)d_in[21];
  const float* bagg = (const float*)d_in[22];
  const float* Wq_t = (const float*)d_in[23];
  const float* Wk_t = (const float*)d_in[24];
  const float* Wv_t = (const float*)d_in[25];
  const float* q_agg = (const float*)d_in[26];
  const float* Wk_tg = (const float*)d_in[27];
  const float* Wv_tg = (const float*)d_in[28];
  const float* Wout = (const float*)d_in[29];
  const float* bout = (const float*)d_in[30];

  const int MX  = BB * TT * NN;   // 50496 rows (mult of 32)
  const int MA0 = BB * TT * N0_;  // 6144
  const int MA1 = BB * TT * N1_;  // 12288

  // workspace bump allocator (assumes ws_size is sufficient, ~350 MB)
  char* p = (char*)d_ws;
  auto alloc = [&](size_t bytes) -> char* {
    char* r = p; p += (bytes + 255) & ~(size_t)255; return r;
  };
  float* lam  = (float*)alloc(4);
  bf16* xb   = (bf16*)alloc((size_t)MX * DD * 2);
  bf16* a0b  = (bf16*)alloc((size_t)MA0 * DD * 2);
  bf16* a1b  = (bf16*)alloc((size_t)MA1 * DD * 2);
  bf16* qgb  = (bf16*)alloc((size_t)NN * SEG_ * 128 * 2);
  bf16* qs   = (bf16*)alloc((size_t)MX * 256 * 2);
  bf16* ks   = (bf16*)alloc((size_t)MX * 256 * 2);
  bf16* vs   = (bf16*)alloc((size_t)MX * 256 * 2);
  bf16* qt   = (bf16*)alloc((size_t)MX * 128 * 2);
  bf16* kt   = (bf16*)alloc((size_t)MX * 128 * 2);
  bf16* vt   = (bf16*)alloc((size_t)MX * 128 * 2);
  bf16* kg   = (bf16*)alloc((size_t)MX * 128 * 2);
  bf16* vg   = (bf16*)alloc((size_t)MX * 128 * 2);
  bf16* qa0  = (bf16*)alloc((size_t)MA0 * 128 * 2);
  bf16* ka0  = (bf16*)alloc((size_t)MA0 * 128 * 2);
  bf16* va0  = (bf16*)alloc((size_t)MA0 * 128 * 2);
  bf16* qa1  = (bf16*)alloc((size_t)MA1 * 128 * 2);
  bf16* ka1  = (bf16*)alloc((size_t)MA1 * 128 * 2);
  bf16* va1  = (bf16*)alloc((size_t)MA1 * 128 * 2);
  float* sx0 = (float*)alloc((size_t)MA0 * 128 * 4);
  float* sx1 = (float*)alloc((size_t)MA1 * 128 * 4);
  float* sgsum = (float*)alloc((size_t)MX * 128 * 4);
  bf16* sgb  = (bf16*)alloc((size_t)MX * 128 * 2);
  float* tgx = (float*)alloc((size_t)BB * NN * SEG_ * 128 * 4);
  bf16* stb  = (bf16*)alloc((size_t)MX * STW * 2);
  bf16* wqs = (bf16*)alloc(256 * 512 * 2);
  bf16* wks = (bf16*)alloc(256 * 512 * 2);
  bf16* wvs = (bf16*)alloc(256 * 512 * 2);
  bf16* wqt = (bf16*)alloc(128 * 512 * 2);
  bf16* wkt = (bf16*)alloc(128 * 512 * 2);
  bf16* wvt = (bf16*)alloc(128 * 512 * 2);
  bf16* wktg = (bf16*)alloc(128 * 512 * 2);
  bf16* wvtg = (bf16*)alloc(128 * 512 * 2);
  bf16* wqa0 = (bf16*)alloc(128 * 512 * 2);
  bf16* wka0 = (bf16*)alloc(128 * 512 * 2);
  bf16* wva0 = (bf16*)alloc(128 * 512 * 2);
  bf16* wqa1 = (bf16*)alloc(128 * 512 * 2);
  bf16* wka1 = (bf16*)alloc(128 * 512 * 2);
  bf16* wva1 = (bf16*)alloc(128 * 512 * 2);
  bf16* wagg = (bf16*)alloc(128 * 128 * 2);
  bf16* wout = (bf16*)alloc(512 * 640 * 2);

  auto cast = [&](const float* src, bf16* dst, size_t n) {
    int blocks = (int)((n + 255) / 256); if (blocks > 8192) blocks = 8192;
    cast_f32_to_bf16<<<blocks, 256, 0, stream>>>(src, dst, n);
  };
  auto gemm = [&](const bf16* A, int lda, const bf16* W, int ldw, const float* bias,
                  void* C, int ldc, int coff, int cbf16, int M, int E, int K) {
    dim3 g(M / 32, E / 64, 1);
    wmma_gemm_kernel<<<g, 32, 0, stream>>>(A, lda, W, ldw, bias, C, ldc, coff, cbf16, M, E, K);
  };

  // 1) casts
  cast(x,   xb,  (size_t)MX * DD);
  cast(ax0, a0b, (size_t)MA0 * DD);
  cast(ax1, a1b, (size_t)MA1 * DD);
  cast(q_agg, qgb, (size_t)NN * SEG_ * 128);
  cast(Wq_s, wqs, 256 * 512); cast(Wk_s, wks, 256 * 512); cast(Wv_s, wvs, 256 * 512);
  cast(Wq_t, wqt, 128 * 512); cast(Wk_t, wkt, 128 * 512); cast(Wv_t, wvt, 128 * 512);
  cast(Wk_tg, wktg, 128 * 512); cast(Wv_tg, wvtg, 128 * 512);
  cast(Wq_a0, wqa0, 128 * 512); cast(Wk_a0, wka0, 128 * 512); cast(Wv_a0, wva0, 128 * 512);
  cast(Wq_a1, wqa1, 128 * 512); cast(Wk_a1, wka1, 128 * 512); cast(Wv_a1, wva1, 128 * 512);
  cast(Wagg, wagg, 128 * 128);
  cast(Wout, wout, 512 * 640);

  // 2) lambda scalar
  lam_kernel<<<1, 32, 0, stream>>>(lq1, lk1, lq2, lk2, lam);

  // 3) projections (WMMA GEMMs, bf16 out)
  gemm(xb, 512, wqs, 512, nullptr, qs, 256, 0, 1, MX, 256, 512);
  gemm(xb, 512, wks, 512, nullptr, ks, 256, 0, 1, MX, 256, 512);
  gemm(xb, 512, wvs, 512, nullptr, vs, 256, 0, 1, MX, 256, 512);
  gemm(xb, 512, wqt, 512, nullptr, qt, 128, 0, 1, MX, 128, 512);
  gemm(xb, 512, wkt, 512, nullptr, kt, 128, 0, 1, MX, 128, 512);
  gemm(xb, 512, wvt, 512, nullptr, vt, 128, 0, 1, MX, 128, 512);
  gemm(xb, 512, wktg, 512, nullptr, kg, 128, 0, 1, MX, 128, 512);
  gemm(xb, 512, wvtg, 512, nullptr, vg, 128, 0, 1, MX, 128, 512);
  gemm(a0b, 512, wqa0, 512, nullptr, qa0, 128, 0, 1, MA0, 128, 512);
  gemm(a0b, 512, wka0, 512, nullptr, ka0, 128, 0, 1, MA0, 128, 512);
  gemm(a0b, 512, wva0, 512, nullptr, va0, 128, 0, 1, MA0, 128, 512);
  gemm(a1b, 512, wqa1, 512, nullptr, qa1, 128, 0, 1, MA1, 128, 512);
  gemm(a1b, 512, wka1, 512, nullptr, ka1, 128, 0, 1, MA1, 128, 512);
  gemm(a1b, 512, wva1, 512, nullptr, va1, 128, 0, 1, MA1, 128, 512);

  // 4) differential spatial attention -> st[:, 0:256)
  diff_attn_kernel<<<dim3(17, 4, BB * TT), 32, 0, stream>>>(qs, ks, vs, lam, ln_g, ln_b, stb);

  // 5) temporal attention -> st[:, 384:512)
  temporal_attn_kernel<<<dim3(NN, 2, BB), 32, 0, stream>>>(qt, kt, vt, stb, 384);

  // 6) temporal-global attention -> tgx, then tmp_map contraction -> st[:, 512:640)
  tglobal_attn_kernel<<<dim3(NN, 2, BB), 32, 0, stream>>>(qgb, kg, vg, tgx);
  {
    size_t total = (size_t)MX * 128;
    tgmap_kernel<<<(int)((total + 255) / 256), 256, 0, stream>>>(tmap, tgx, stb);
  }

  // 7) aggregation branches -> sx0/sx1 -> M-mix -> Wagg GEMM -> st[:, 256:384)
  agg_attn_kernel<<<dim3(N0_ / 16, 2, BB * TT), 32, 0, stream>>>(qa0, ka0, va0, sx0, N0_);
  agg_attn_kernel<<<dim3(N1_ / 16, 2, BB * TT), 32, 0, stream>>>(qa1, ka1, va1, sx1, N1_);
  {
    size_t total = (size_t)MX * 128;
    mproj_kernel<<<(int)((total + 255) / 256), 256, 0, stream>>>(sx0, M0w, sx1, M1w, sgsum);
    cast(sgsum, sgb, total);
  }
  gemm(sgb, 128, wagg, 128, bagg, stb, STW, 256, 1, MX, 128, 128);

  // 8) final projection: d_out = st @ Wout^T + bout (fp32 out)
  gemm(stb, STW, wout, STW, bout, d_out, 512, 0, 0, MX, 512, STW);
}
